// DPCA2D_62878321213854
// MI455X (gfx1250) — matrix operand order repaired and checked by reading, not compiled, not run on hardware
//
#include <hip/hip_runtime.h>

typedef __attribute__((ext_vector_type(16))) _Float16 v16h;
typedef __attribute__((ext_vector_type(8)))  float    v8f;

#define DIMC 256
#define HEADS 8
#define INNER 512
#define HW 4096          // 64*64 pixels per batch
#define NPIX 16384       // 4 batches * HW
#define NKEY 1024        // 32*32 pruned keys

// ---------------- WMMA helpers (CDNA5 layouts, wave32) ----------------

static __device__ __forceinline__ v8f wmma_f16(v16h a, v16h b, v8f c) {
    return __builtin_amdgcn_wmma_f32_16x16x32_f16(false, a, false, b, (short)0, c, false, false);
}

// A matrix 16x32 f16, row-major source with row stride ld.
// lane<16: row=lane, K 0..15 (elts 0-15)
// lane>=16: row=lane-16, K 16..31
static __device__ __forceinline__ v16h ld_a16x32(const _Float16* __restrict__ src, int ld) {
    int lane = threadIdx.x & 31;
    int m = lane & 15;
    int kb = (lane < 16) ? 0 : 16;
    const _Float16* r = src + (size_t)m * ld;
    v16h a;
#pragma unroll
    for (int i = 0; i < 8; ++i) a[i] = r[kb + i];
#pragma unroll
    for (int i = 0; i < 8; ++i) a[8 + i] = r[kb + 8 + i];
    return a;
}

// B = W^T where W is [N x K] row-major (K-chunk of 32 starting at w).
// lane&15 = column n, lane>=16 covers K 16..31. Contiguous loads.
static __device__ __forceinline__ v16h ld_bt16x32(const _Float16* __restrict__ w, int ldw) {
    int lane = threadIdx.x & 31;
    int n = lane & 15;
    int kb = (lane < 16) ? 0 : 16;
    const _Float16* r = w + (size_t)n * ldw + kb;
    v16h b;
#pragma unroll
    for (int i = 0; i < 16; ++i) b[i] = r[i];
    return b;
}

// ---------------- Stage 1: channel LayerNorm + transpose to [pixel][chan] f16 ----------------

__global__ void ln_pack_kernel(const float* __restrict__ x, const float* __restrict__ g,
                               const float* __restrict__ bb, _Float16* __restrict__ dst) {
    int t = blockIdx.x * blockDim.x + threadIdx.x;   // 0..NPIX-1
    int b = t >> 12, p = t & 4095;
    const float* xp = x + (size_t)b * DIMC * HW + p; // NCHW, channel stride HW (coalesced over p)
    float s = 0.f, s2 = 0.f;
    for (int c = 0; c < DIMC; ++c) { float v = xp[(size_t)c * HW]; s += v; s2 += v * v; }
    float mean = s * (1.0f / DIMC);
    float var  = s2 * (1.0f / DIMC) - mean * mean;
    float inv  = rsqrtf(var + 1e-5f);
    _Float16* d = dst + (size_t)t * DIMC;
    for (int c = 0; c < DIMC; ++c) {
        float v = (xp[(size_t)c * HW] - mean) * inv * g[c] + bb[c];
        d[c] = (_Float16)v;
    }
}

__global__ void f32_to_f16_kernel(const float* __restrict__ src, _Float16* __restrict__ dst, int n) {
    int i = blockIdx.x * blockDim.x + threadIdx.x;
    if (i < n) dst[i] = (_Float16)src[i];
}

// ---------------- Stage 2: pipelined WMMA GEMM  C[M x N] = A[M x K] * W[N x K]^T ----------------
// One wave = 32x64 output tile. K is a template constant: full unroll turns the
// double-buffer rotation into SSA renames (no v_mov copies, no WMMA hazard nops)
// and lets the scheduler keep each next chunk's loads in flight under the WMMAs.

template <int K>
__global__ __launch_bounds__(32) void gemm32_kernel(const _Float16* __restrict__ A,
                                                    const _Float16* __restrict__ W,
                                                    _Float16* __restrict__ C, int N) {
    int row0 = blockIdx.x * 32;
    int col0 = blockIdx.y * 64;
    int lane = threadIdx.x;
    const _Float16* Ar0 = A + (size_t)row0 * K;
    const _Float16* Ar1 = A + (size_t)(row0 + 16) * K;
    const _Float16* W0 = W + (size_t)(col0 + 0)  * K;
    const _Float16* W1 = W + (size_t)(col0 + 16) * K;
    const _Float16* W2 = W + (size_t)(col0 + 32) * K;
    const _Float16* W3 = W + (size_t)(col0 + 48) * K;

    v8f acc0[4] = {}, acc1[4] = {};
    v16h a0 = ld_a16x32(Ar0, K);
    v16h a1 = ld_a16x32(Ar1, K);
    v16h b0 = ld_bt16x32(W0, K);
    v16h b1 = ld_bt16x32(W1, K);
    v16h b2 = ld_bt16x32(W2, K);
    v16h b3 = ld_bt16x32(W3, K);

#pragma unroll
    for (int kc = 0; kc < K; kc += 32) {
        const int kn = kc + 32;
        v16h a0n, a1n, b0n, b1n, b2n, b3n;
        if (kn < K) {           // compile-time: issue next chunk's loads first
            a0n = ld_a16x32(Ar0 + kn, K);
            a1n = ld_a16x32(Ar1 + kn, K);
            b0n = ld_bt16x32(W0 + kn, K);
            b1n = ld_bt16x32(W1 + kn, K);
            b2n = ld_bt16x32(W2 + kn, K);
            b3n = ld_bt16x32(W3 + kn, K);
        }
        acc0[0] = wmma_f16(a0, b0, acc0[0]);
        acc1[0] = wmma_f16(a1, b0, acc1[0]);
        acc0[1] = wmma_f16(a0, b1, acc0[1]);
        acc1[1] = wmma_f16(a1, b1, acc1[1]);
        acc0[2] = wmma_f16(a0, b2, acc0[2]);
        acc1[2] = wmma_f16(a1, b2, acc1[2]);
        acc0[3] = wmma_f16(a0, b3, acc0[3]);
        acc1[3] = wmma_f16(a1, b3, acc1[3]);
        if (kn < K) { a0 = a0n; a1 = a1n; b0 = b0n; b1 = b1n; b2 = b2n; b3 = b3n; }
    }

    int rowoff = (lane < 16) ? 0 : 8;
    int col = lane & 15;
#pragma unroll
    for (int t = 0; t < 4; ++t)
#pragma unroll
        for (int r = 0; r < 8; ++r) {
            C[(size_t)(row0 + r + rowoff) * N + col0 + t * 16 + col]      = (_Float16)acc0[t][r];
            C[(size_t)(row0 + 16 + r + rowoff) * N + col0 + t * 16 + col] = (_Float16)acc1[t][r];
        }
}

// ---------------- Stage 3: per-(pixel,head) L2 norm over 64 channels ----------------

__global__ void l2n_kernel(_Float16* __restrict__ base, int stridePix, int n) {
    int t = blockIdx.x * blockDim.x + threadIdx.x;
    if (t >= n) return;
    int pixel = t >> 3, head = t & 7;
    _Float16* p = base + (size_t)pixel * stridePix + head * 64;
    float s = 0.f;
    for (int d = 0; d < 64; ++d) { float v = (float)p[d]; s += v * v; }
    float inv = 1.0f / fmaxf(sqrtf(s), 1e-12f);
    for (int d = 0; d < 64; ++d) p[d] = (_Float16)((float)p[d] * inv);
}

// ---------------- Stage 4: probes for pruning ----------------

__global__ void qprobe_kernel(const _Float16* __restrict__ qT, float* __restrict__ qprobe) {
    int bh = blockIdx.x, d = threadIdx.x;          // 32 blocks x 64 threads
    int b = bh >> 3, h = bh & 7;
    const _Float16* p = qT + (size_t)b * HW * INNER + h * 64 + d;
    float s = 0.f;
    for (int i = 0; i < HW; ++i) s += fabsf((float)p[(size_t)i * INNER]);
    qprobe[bh * 64 + d] = s;
}

__global__ void kstats_kernel(const _Float16* __restrict__ kvT, float* __restrict__ krow,
                              float* __restrict__ kcol) {
    int rc = blockIdx.x;                            // row or col index 0..63
    int bh = blockIdx.y;
    int d = threadIdx.x;
    int b = bh >> 3, h = bh & 7;
    const _Float16* base = kvT + (size_t)b * HW * 1024 + h * 64 + d;
    float sr = 0.f, sc = 0.f;
    for (int i = 0; i < 64; ++i) {
        sr += fabsf((float)base[(size_t)(rc * 64 + i) * 1024]);   // row rc summed over cols
        sc += fabsf((float)base[(size_t)(i * 64 + rc) * 1024]);   // col rc summed over rows
    }
    krow[(bh * 64 + rc) * 64 + d] = sr;
    kcol[(bh * 64 + rc) * 64 + d] = sc;
}

// top-32-of-64 via rank counting (selection set is order-invariant post-softmax)
__global__ void topk_kernel(const float* __restrict__ qprobe, const float* __restrict__ krow,
                            const float* __restrict__ kcol, int* __restrict__ hsel,
                            int* __restrict__ wsel) {
    __shared__ float sc[64];
    __shared__ int fl[64];
    int h = threadIdx.x, bh = blockIdx.x;
    float s = 0.f;
    for (int d = 0; d < 64; ++d) s += qprobe[bh * 64 + d] * krow[(bh * 64 + h) * 64 + d];
    sc[h] = s; __syncthreads();
    int rank = 0;
    for (int j = 0; j < 64; ++j) rank += (sc[j] > s) || (sc[j] == s && j < h);
    fl[h] = (rank < 32); __syncthreads();
    if (fl[h]) { int pos = 0; for (int j = 0; j < h; ++j) pos += fl[j]; hsel[bh * 32 + pos] = h; }
    __syncthreads();
    s = 0.f;
    for (int d = 0; d < 64; ++d) s += qprobe[bh * 64 + d] * kcol[(bh * 64 + h) * 64 + d];
    sc[h] = s; __syncthreads();
    rank = 0;
    for (int j = 0; j < 64; ++j) rank += (sc[j] > s) || (sc[j] == s && j < h);
    fl[h] = (rank < 32); __syncthreads();
    if (fl[h]) { int pos = 0; for (int j = 0; j < h; ++j) pos += fl[j]; wsel[bh * 32 + pos] = h; }
}

// K kept row-major [bh][key][64]; V stored TRANSPOSED [bh][d=64][key=1024] so
// both S=Q*K^T and O+=P*V fragment loads are contiguous (vectorize to b128).
__global__ void gather_kernel(const _Float16* __restrict__ kvT, const int* __restrict__ hsel,
                              const int* __restrict__ wsel, _Float16* __restrict__ kf,
                              _Float16* __restrict__ vfT) {
    int j = blockIdx.x, bh = blockIdx.y, d = threadIdx.x;
    int b = bh >> 3, h = bh & 7;
    int hi = hsel[bh * 32 + (j >> 5)];
    int wi = wsel[bh * 32 + (j & 31)];
    size_t src = ((size_t)b * HW + hi * 64 + wi) * 1024 + h * 64 + d;
    kf [((size_t)bh * NKEY + j) * 64 + d]  = kvT[src];          // k = kv channels [0,512)
    vfT[((size_t)bh * 64 + d) * NKEY + j]  = kvT[src + 512];    // v = kv channels [512,1024)
}

// ---------------- Stage 5: fused flash attention (1 wave = 16 queries x all 1024 keys) ----------------

__global__ __launch_bounds__(32) void flash_kernel(const _Float16* __restrict__ qT,
                                                   const _Float16* __restrict__ kf,
                                                   const _Float16* __restrict__ vfT,
                                                   _Float16* __restrict__ aoT) {
    int qt = blockIdx.x;                 // 0..255 query tiles of 16
    int bh = blockIdx.y;                 // 0..31
    int b = bh >> 3, h = bh & 7;
    int lane = threadIdx.x;
    __shared__ float    sS[16 * 32];
    __shared__ _Float16 sP[16 * 32];
    __shared__ float    sM[16], sL[16], sSc[16];

    const _Float16* qbase = qT + ((size_t)b * HW + qt * 16) * INNER + h * 64;
    v16h aq0 = ld_a16x32(qbase, INNER);        // channels 0..31
    v16h aq1 = ld_a16x32(qbase + 32, INNER);   // channels 32..63

    if (lane < 16) { sM[lane] = -1e30f; sL[lane] = 0.f; }
    __syncthreads();

    v8f o0 = {}, o1 = {}, o2 = {}, o3 = {};
    int rowoff = (lane < 16) ? 0 : 8;
    int col = lane & 15;
    const _Float16* vbase = vfT + (size_t)bh * 64 * NKEY;

    for (int j = 0; j < 32; ++j) {              // 32 chunks of 32 keys
        const _Float16* kp = kf + ((size_t)bh * NKEY + j * 32) * 64;
        // K fragments (contiguous): issue all four before the S WMMAs
        v16h kb0 = ld_bt16x32(kp, 64);
        v16h kb1 = ld_bt16x32(kp + 32, 64);
        v16h kb2 = ld_bt16x32(kp + 16 * 64, 64);
        v16h kb3 = ld_bt16x32(kp + 16 * 64 + 32, 64);
        // V fragments (transposed layout, contiguous): issue now, consumed only
        // after the softmax section -> latency hidden under LDS + VALU work.
        v16h vb0 = ld_bt16x32(vbase + (size_t)(0)  * NKEY + j * 32, NKEY);
        v16h vb1 = ld_bt16x32(vbase + (size_t)(16) * NKEY + j * 32, NKEY);
        v16h vb2 = ld_bt16x32(vbase + (size_t)(32) * NKEY + j * 32, NKEY);
        v16h vb3 = ld_bt16x32(vbase + (size_t)(48) * NKEY + j * 32, NKEY);

        v8f s0 = {}, s1 = {};
        s0 = wmma_f16(aq0, kb0, s0);
        s0 = wmma_f16(aq1, kb1, s0);
        s1 = wmma_f16(aq0, kb2, s1);
        s1 = wmma_f16(aq1, kb3, s1);
#pragma unroll
        for (int r = 0; r < 8; ++r) {
            sS[(r + rowoff) * 32 + col]      = s0[r];
            sS[(r + rowoff) * 32 + 16 + col] = s1[r];
        }
        __syncthreads();
        if (lane < 16) {                        // online softmax, one row per lane
            int row = lane;
            float mold = sM[row], mx = mold;
            for (int c = 0; c < 32; ++c) mx = fmaxf(mx, sS[row * 32 + c]);
            float scale = __expf(mold - mx);
            float l = sL[row] * scale;
            for (int c = 0; c < 32; ++c) {
                float e = __expf(sS[row * 32 + c] - mx);
                l += e;
                sP[row * 32 + c] = (_Float16)e;
            }
            sM[row] = mx; sL[row] = l; sSc[row] = scale;
        }
        __syncthreads();
        float scr[8];
#pragma unroll
        for (int r = 0; r < 8; ++r) scr[r] = sSc[r + rowoff];
#pragma unroll
        for (int r = 0; r < 8; ++r) { o0[r] *= scr[r]; o1[r] *= scr[r]; o2[r] *= scr[r]; o3[r] *= scr[r]; }
        v16h ap = ld_a16x32(sP, 32);            // P: 16 queries x 32 keys
        o0 = wmma_f16(ap, vb0, o0);
        o1 = wmma_f16(ap, vb1, o1);
        o2 = wmma_f16(ap, vb2, o2);
        o3 = wmma_f16(ap, vb3, o3);
        __syncthreads();
    }
    float linv[8];
#pragma unroll
    for (int r = 0; r < 8; ++r) linv[r] = 1.0f / sL[r + rowoff];
    _Float16* out = aoT + ((size_t)b * HW + qt * 16) * INNER + h * 64;
#pragma unroll
    for (int r = 0; r < 8; ++r) {
        _Float16* orow = out + (size_t)(r + rowoff) * INNER;
        orow[col]      = (_Float16)(o0[r] * linv[r]);
        orow[16 + col] = (_Float16)(o1[r] * linv[r]);
        orow[32 + col] = (_Float16)(o2[r] * linv[r]);
        orow[48 + col] = (_Float16)(o3[r] * linv[r]);
    }
}

// ---------------- Stage 6: out-proj GEMM + gamma*x + residual, transposed NCHW store ----------------

__global__ __launch_bounds__(32) void outgemm_kernel(const _Float16* __restrict__ A,
                                                     const _Float16* __restrict__ W,
                                                     const float* __restrict__ qsrc,
                                                     const float* __restrict__ gamma,
                                                     float* __restrict__ out) {
    int row0 = blockIdx.x * 32;     // pixel rows
    int col0 = blockIdx.y * 64;     // output channels
    int lane = threadIdx.x;
    const _Float16* Ar0 = A + (size_t)row0 * INNER;
    const _Float16* Ar1 = A + (size_t)(row0 + 16) * INNER;
    const _Float16* W0 = W + (size_t)(col0 + 0)  * INNER;
    const _Float16* W1 = W + (size_t)(col0 + 16) * INNER;
    const _Float16* W2 = W + (size_t)(col0 + 32) * INNER;
    const _Float16* W3 = W + (size_t)(col0 + 48) * INNER;

    v8f acc0[4] = {}, acc1[4] = {};
    v16h a0 = ld_a16x32(Ar0, INNER);
    v16h a1 = ld_a16x32(Ar1, INNER);
    v16h b0 = ld_bt16x32(W0, INNER);
    v16h b1 = ld_bt16x32(W1, INNER);
    v16h b2 = ld_bt16x32(W2, INNER);
    v16h b3 = ld_bt16x32(W3, INNER);

#pragma unroll
    for (int kc = 0; kc < INNER; kc += 32) {
        const int kn = kc + 32;
        v16h a0n, a1n, b0n, b1n, b2n, b3n;
        if (kn < INNER) {
            a0n = ld_a16x32(Ar0 + kn, INNER);
            a1n = ld_a16x32(Ar1 + kn, INNER);
            b0n = ld_bt16x32(W0 + kn, INNER);
            b1n = ld_bt16x32(W1 + kn, INNER);
            b2n = ld_bt16x32(W2 + kn, INNER);
            b3n = ld_bt16x32(W3 + kn, INNER);
        }
        acc0[0] = wmma_f16(a0, b0, acc0[0]);
        acc1[0] = wmma_f16(a1, b0, acc1[0]);
        acc0[1] = wmma_f16(a0, b1, acc0[1]);
        acc1[1] = wmma_f16(a1, b1, acc1[1]);
        acc0[2] = wmma_f16(a0, b2, acc0[2]);
        acc1[2] = wmma_f16(a1, b2, acc1[2]);
        acc0[3] = wmma_f16(a0, b3, acc0[3]);
        acc1[3] = wmma_f16(a1, b3, acc1[3]);
        if (kn < INNER) { a0 = a0n; a1 = a1n; b0 = b0n; b1 = b1n; b2 = b2n; b3 = b3n; }
    }

    float g0 = gamma[0];
    int rowoff = (lane < 16) ? 0 : 8;
    int col = lane & 15;
    int b = row0 >> 12;
#pragma unroll
    for (int t = 0; t < 4; ++t)
#pragma unroll
        for (int r = 0; r < 8; ++r) {
            int o = col0 + t * 16 + col;
            int p0 = (row0 + r + rowoff) & 4095;
            int p1 = (row0 + 16 + r + rowoff) & 4095;
            size_t idx0 = ((size_t)b * DIMC + o) * HW + p0;
            size_t idx1 = ((size_t)b * DIMC + o) * HW + p1;
            out[idx0] = g0 * acc0[t][r] + qsrc[idx0];
            out[idx1] = g0 * acc1[t][r] + qsrc[idx1];
        }
}

// ---------------- Host launch ----------------

extern "C" void kernel_launch(void* const* d_in, const int* in_sizes, int n_in,
                              void* d_out, int out_size, void* d_ws, size_t ws_size,
                              hipStream_t stream) {
    (void)in_sizes; (void)n_in; (void)out_size; (void)ws_size;
    const float* context = (const float*)d_in[0];
    const float* qsrc    = (const float*)d_in[1];
    const float* g_ctx   = (const float*)d_in[2];
    const float* b_ctx   = (const float*)d_in[3];
    const float* g_qs    = (const float*)d_in[4];
    const float* b_qs    = (const float*)d_in[5];
    const float* w_kv    = (const float*)d_in[6];
    const float* w_q     = (const float*)d_in[7];
    const float* w_out   = (const float*)d_in[8];
    const float* gamma   = (const float*)d_in[9];
    float* out = (float*)d_out;

    char* ws = (char*)d_ws;
    size_t off = 0;
    auto alloc = [&](size_t bytes) -> void* {
        void* p = (void*)(ws + off);
        off += (bytes + 255) & ~(size_t)255;
        return p;
    };
    // ~91 MB total workspace
    _Float16* ctxT  = (_Float16*)alloc((size_t)NPIX * DIMC * 2);    // 8 MB
    _Float16* qsT   = (_Float16*)alloc((size_t)NPIX * DIMC * 2);    // 8 MB
    _Float16* kvT   = (_Float16*)alloc((size_t)NPIX * 1024 * 2);    // 32 MB
    _Float16* qTb   = (_Float16*)alloc((size_t)NPIX * INNER * 2);   // 16 MB
    _Float16* aoT   = (_Float16*)alloc((size_t)NPIX * INNER * 2);   // 16 MB
    _Float16* wkv16 = (_Float16*)alloc((size_t)1024 * DIMC * 2);
    _Float16* wq16  = (_Float16*)alloc((size_t)INNER * DIMC * 2);
    _Float16* wo16  = (_Float16*)alloc((size_t)DIMC * INNER * 2);
    float* qprobe   = (float*)alloc(32 * 64 * 4);
    float* krow     = (float*)alloc(32 * 64 * 64 * 4);
    float* kcol     = (float*)alloc(32 * 64 * 64 * 4);
    int*   hsel     = (int*)alloc(32 * 32 * 4);
    int*   wsel     = (int*)alloc(32 * 32 * 4);
    _Float16* kfb   = (_Float16*)alloc((size_t)32 * NKEY * 64 * 2); // 4 MB
    _Float16* vfbT  = (_Float16*)alloc((size_t)32 * 64 * NKEY * 2); // 4 MB (transposed)

    // 1. LayerNorm + transpose + f16 pack
    ln_pack_kernel<<<NPIX / 256, 256, 0, stream>>>(context, g_ctx, b_ctx, ctxT);
    ln_pack_kernel<<<NPIX / 256, 256, 0, stream>>>(qsrc, g_qs, b_qs, qsT);
    f32_to_f16_kernel<<<(1024 * DIMC + 255) / 256, 256, 0, stream>>>(w_kv, wkv16, 1024 * DIMC);
    f32_to_f16_kernel<<<(INNER * DIMC + 255) / 256, 256, 0, stream>>>(w_q, wq16, INNER * DIMC);
    f32_to_f16_kernel<<<(DIMC * INNER + 255) / 256, 256, 0, stream>>>(w_out, wo16, DIMC * INNER);
    // 2. kv / q projections (fully-unrolled pipelined WMMA, K=256)
    gemm32_kernel<DIMC><<<dim3(NPIX / 32, 1024 / 64), 32, 0, stream>>>(ctxT, wkv16, kvT, 1024);
    gemm32_kernel<DIMC><<<dim3(NPIX / 32, INNER / 64), 32, 0, stream>>>(qsT, wq16, qTb, INNER);
    // 3. l2 normalize q and k per (pixel, head)
    l2n_kernel<<<(NPIX * HEADS + 255) / 256, 256, 0, stream>>>(qTb, INNER, NPIX * HEADS);
    l2n_kernel<<<(NPIX * HEADS + 255) / 256, 256, 0, stream>>>(kvT, 1024, NPIX * HEADS);
    // 4. pruning probes + top-k + gather (V transposed for contiguous fragment loads)
    qprobe_kernel<<<32, 64, 0, stream>>>(qTb, qprobe);
    kstats_kernel<<<dim3(64, 32), 64, 0, stream>>>(kvT, krow, kcol);
    topk_kernel<<<32, 64, 0, stream>>>(qprobe, krow, kcol, hsel, wsel);
    gather_kernel<<<dim3(1024, 32), 64, 0, stream>>>(kvT, hsel, wsel, kfb, vfbT);
    // 5. fused flash attention (WMMA)
    flash_kernel<<<dim3(256, 32), 32, 0, stream>>>(qTb, kfb, vfbT, aoT);
    // 6. output projection + residual (fully-unrolled pipelined WMMA, K=512)
    outgemm_kernel<<<dim3(NPIX / 32, DIMC / 64), 32, 0, stream>>>(aoT, wo16, qsrc, gamma, out);
}
